// LinearAttention_88390426952202
// MI455X (gfx1250) — matrix operand order repaired and checked by reading, not compiled, not run on hardware
//
#include <hip/hip_runtime.h>
#include <hip/hip_bf16.h>

typedef __bf16 bf16_t;
typedef __attribute__((ext_vector_type(16))) __bf16 bf16x16;
typedef __attribute__((ext_vector_type(8)))  __bf16 bf16x8;
typedef __attribute__((ext_vector_type(4)))  __bf16 bf16x4;
typedef __attribute__((ext_vector_type(8)))  float  f32x8;

#define D_MODEL 1024
#define N_HEADS 16
#define D_HEAD  64
#define BATCH   4
#define SEQ     4096
#define BH      (BATCH * N_HEADS)
#define CHUNK   64
#define NCHUNK  (SEQ / CHUNK)

static __device__ __forceinline__ f32x8 zero8() {
  f32x8 v = {0.f, 0.f, 0.f, 0.f, 0.f, 0.f, 0.f, 0.f};
  return v;
}

static __device__ __forceinline__ f32x8 wmma_bf16(bf16x16 a, bf16x16 b, f32x8 c) {
  return __builtin_amdgcn_wmma_f32_16x16x32_bf16(false, a, false, b, (short)0, c,
                                                 false, false);
}

// A fragment (16x32 bf16, MxK): lane = row m (0..15), per-lane K elements
// {k0..k0+7, k0+16..k0+23}, k0 = (lane>>4)*8.  `base` points at tile origin,
// rows are `ld` elements apart, K contiguous.
static __device__ __forceinline__ bf16x16 frag_a(const bf16_t* base, int ld, int lane) {
  const int row = lane & 15;
  const int k0  = (lane >> 4) << 3;
  const bf16_t* p = base + row * ld + k0;
  bf16x8 lo = *(const bf16x8*)(p);
  bf16x8 hi = *(const bf16x8*)(p + 16);
  bf16x16 f;
#pragma unroll
  for (int i = 0; i < 8; ++i) { f[i] = lo[i]; f[i + 8] = hi[i]; }
  return f;
}

// B fragment (32x16 bf16, KxN): lane = col n (0..15), per-lane K elements
// {k0..k0+7, k0+16..k0+23}, k0 = (lane>>4)*8.  Storage is B^T: `base` points at tile origin of a
// (n-major, K-contiguous) buffer with row stride `ld`.
static __device__ __forceinline__ bf16x16 frag_b(const bf16_t* base, int ld, int lane) {
  const int col = lane & 15;
  const int k0  = (lane >> 4) << 3;
  const bf16_t* p = base + col * ld + k0;
  bf16x8 lo = *(const bf16x8*)(p);
  bf16x8 hi = *(const bf16x8*)(p + 16);
  bf16x16 f;
#pragma unroll
  for (int i = 0; i < 8; ++i) { f[i] = lo[i]; f[i + 8] = hi[i]; }
  return f;
}

// ---------------------------------------------------------------- fp32 -> bf16
__global__ __launch_bounds__(256) void cvt_bf16_k(const float* __restrict__ in,
                                                  bf16_t* __restrict__ out, int n4) {
  int i = blockIdx.x * blockDim.x + threadIdx.x;
  if (i < n4) {
    float4 v = ((const float4*)in)[i];
    bf16x4 r;
    r[0] = (bf16_t)v.x; r[1] = (bf16_t)v.y; r[2] = (bf16_t)v.z; r[3] = (bf16_t)v.w;
    ((bf16x4*)out)[i] = r;
  }
}

// ------------------------------------------------------------- TN WMMA GEMM
// C[m,n] = sum_k A[m,k] * B[n,k]   (A: MxK row-major, B: NxK row-major)
// EPI==0: store fp32 C row-major.
// EPI==1: QKV epilogue -> apply phi=elu+1 to q/k, scatter bf16 to (b,h,l,d).
template <int EPI>
__global__ __launch_bounds__(256) void gemm_bf16_k(
    const bf16_t* __restrict__ A, const bf16_t* __restrict__ B,
    float* __restrict__ C,
    bf16_t* __restrict__ qo, bf16_t* __restrict__ ko, bf16_t* __restrict__ vo,
    int M, int N, int K) {
  constexpr int LDT = 40;  // 32 + 8 pad, keeps 16B alignment, dodges bank conflicts
  __shared__ __align__(16) bf16_t As[128 * LDT];
  __shared__ __align__(16) bf16_t Bs[128 * LDT];

  const int tid  = threadIdx.x;
  const int lane = tid & 31;
  const int wave = tid >> 5;          // 8 waves (wave32)
  const int wm = wave & 1;            // 2 row groups of 64
  const int wn = wave >> 1;           // 4 col groups of 32
  const int m0 = blockIdx.y * 128;
  const int n0 = blockIdx.x * 128;

  f32x8 acc[4][2];
#pragma unroll
  for (int a = 0; a < 4; ++a)
#pragma unroll
    for (int c = 0; c < 2; ++c) acc[a][c] = zero8();

  const int lr = tid >> 2;            // 0..63
  const int lk = (tid & 3) << 3;      // 0,8,16,24

  for (int kb0 = 0; kb0 < K; kb0 += 32) {
    __syncthreads();
    *(bf16x8*)&As[lr * LDT + lk]        = *(const bf16x8*)&A[(size_t)(m0 + lr) * K + kb0 + lk];
    *(bf16x8*)&As[(lr + 64) * LDT + lk] = *(const bf16x8*)&A[(size_t)(m0 + lr + 64) * K + kb0 + lk];
    *(bf16x8*)&Bs[lr * LDT + lk]        = *(const bf16x8*)&B[(size_t)(n0 + lr) * K + kb0 + lk];
    *(bf16x8*)&Bs[(lr + 64) * LDT + lk] = *(const bf16x8*)&B[(size_t)(n0 + lr + 64) * K + kb0 + lk];
    __syncthreads();

    bf16x16 af[4], bfr[2];
#pragma unroll
    for (int rt = 0; rt < 4; ++rt)
      af[rt] = frag_a(&As[(wm * 64 + rt * 16) * LDT], LDT, lane);
#pragma unroll
    for (int ct = 0; ct < 2; ++ct)
      bfr[ct] = frag_b(&Bs[(wn * 32 + ct * 16) * LDT], LDT, lane);
#pragma unroll
    for (int rt = 0; rt < 4; ++rt)
#pragma unroll
      for (int ct = 0; ct < 2; ++ct)
        acc[rt][ct] = wmma_bf16(af[rt], bfr[ct], acc[rt][ct]);
  }

  const int nL = lane & 15;
  const int mH = (lane >> 4) << 3;
#pragma unroll
  for (int rt = 0; rt < 4; ++rt) {
#pragma unroll
    for (int ct = 0; ct < 2; ++ct) {
      const int mBase = m0 + wm * 64 + rt * 16 + mH;
      const int n     = n0 + wn * 32 + ct * 16 + nL;
#pragma unroll
      for (int r = 0; r < 8; ++r) {
        const float v = acc[rt][ct][r];
        const int m = mBase + r;
        if (EPI == 0) {
          C[(size_t)m * N + n] = v;
        } else {
          const int b = m >> 12, l = m & 4095;           // SEQ = 4096
          const int comp = n >> 10, hd = n & 1023;       // D_MODEL = 1024
          const int h = hd >> 6, d = hd & 63;
          const size_t idx = (((size_t)(b * N_HEADS + h)) * SEQ + l) * D_HEAD + d;
          if (comp == 2) {
            vo[idx] = (bf16_t)v;
          } else {
            const float p = v > 0.f ? v + 1.f : __expf(v);  // elu(x)+1
            if (comp == 0) qo[idx] = (bf16_t)p; else ko[idx] = (bf16_t)p;
          }
        }
      }
    }
  }
}

// -------------------------------------------------- pass 1: per-chunk states
// S_c[d,e] = sum_i K[i,d]*V[i,e];  z_c[d] = sum_i K[i,d]
__global__ __launch_bounds__(128) void chunk_state_k(
    const bf16_t* __restrict__ kb, const bf16_t* __restrict__ vb,
    float* __restrict__ Sloc, float* __restrict__ zloc) {
  const int bh = blockIdx.y, ch = blockIdx.x;
  const bf16_t* Kc = kb + (((size_t)bh * SEQ) + (size_t)ch * CHUNK) * D_HEAD;
  const bf16_t* Vc = vb + (((size_t)bh * SEQ) + (size_t)ch * CHUNK) * D_HEAD;
  __shared__ __align__(16) bf16_t Kt[64 * 72];  // Kt[d][i]
  __shared__ __align__(16) bf16_t Vt[64 * 72];  // Vt[e][i]
  const int tid = threadIdx.x, lane = tid & 31, wave = tid >> 5;
  {
    const int i = tid >> 1, c0 = (tid & 1) * 32;
#pragma unroll 8
    for (int d = 0; d < 32; ++d) {
      Kt[(c0 + d) * 72 + i] = Kc[i * D_HEAD + c0 + d];
      Vt[(c0 + d) * 72 + i] = Vc[i * D_HEAD + c0 + d];
    }
  }
  __syncthreads();

  const int wd = wave >> 1, we = wave & 1;  // 2x2 waves, each owns 32x32 of S
  f32x8 acc[2][2];
#pragma unroll
  for (int a = 0; a < 2; ++a)
#pragma unroll
    for (int c = 0; c < 2; ++c) acc[a][c] = zero8();

  for (int i0 = 0; i0 < 64; i0 += 32) {
    bf16x16 af[2], bfr[2];
#pragma unroll
    for (int rt = 0; rt < 2; ++rt)
      af[rt] = frag_a(&Kt[(wd * 32 + rt * 16) * 72 + i0], 72, lane);
#pragma unroll
    for (int ct = 0; ct < 2; ++ct)
      bfr[ct] = frag_b(&Vt[(we * 32 + ct * 16) * 72 + i0], 72, lane);
#pragma unroll
    for (int rt = 0; rt < 2; ++rt)
#pragma unroll
      for (int ct = 0; ct < 2; ++ct)
        acc[rt][ct] = wmma_bf16(af[rt], bfr[ct], acc[rt][ct]);
  }

  float* Sout = Sloc + ((size_t)bh * NCHUNK + ch) * 4096;
  const int nL = lane & 15, mH = (lane >> 4) << 3;
#pragma unroll
  for (int rt = 0; rt < 2; ++rt)
#pragma unroll
    for (int ct = 0; ct < 2; ++ct)
#pragma unroll
      for (int r = 0; r < 8; ++r)
        Sout[(size_t)(wd * 32 + rt * 16 + mH + r) * 64 + (we * 32 + ct * 16 + nL)] =
            acc[rt][ct][r];

  if (tid < 64) {
    float s = 0.f;
    for (int i = 0; i < 64; ++i) s += (float)Kt[tid * 72 + i];
    zloc[((size_t)bh * NCHUNK + ch) * 64 + tid] = s;
  }
}

// --------------------------------- pass 2: exclusive prefix over chunk states
// Writes S_prefix transposed ([e][d]) as bf16 (ready as WMMA B operand) and
// z_prefix fp32. Running sums stay fp32 in registers.
__global__ __launch_bounds__(256) void prefix_k(
    const float* __restrict__ Sloc, const float* __restrict__ zloc,
    bf16_t* __restrict__ Spt, float* __restrict__ zp) {
  const int bh = blockIdx.x, tid = threadIdx.x;
  float run[16];
#pragma unroll
  for (int j = 0; j < 16; ++j) run[j] = 0.f;
  float runz = 0.f;
  for (int ch = 0; ch < NCHUNK; ++ch) {
    const float* Sc = Sloc + ((size_t)bh * NCHUNK + ch) * 4096;
    bf16_t* So = Spt + ((size_t)bh * NCHUNK + ch) * 4096;
#pragma unroll
    for (int j = 0; j < 16; ++j) {
      const int idx = tid * 16 + j;
      const int d = idx >> 6, e = idx & 63;
      So[e * 64 + d] = (bf16_t)run[j];   // exclusive, transposed
      run[j] += Sc[idx];
    }
    if (tid < 64) {
      const size_t zi = ((size_t)bh * NCHUNK + ch) * 64 + tid;
      zp[zi] = runz;                     // exclusive
      runz += zloc[zi];
    }
  }
}

// ---------------------------------------------------- pass 3: chunk outputs
// num = tril(Q K^T) @ V + Q @ S_prefix ;  den = rowsum(tril(QK^T)) + q.z_prefix
__global__ __launch_bounds__(128) void chunk_out_k(
    const bf16_t* __restrict__ qb, const bf16_t* __restrict__ kb,
    const bf16_t* __restrict__ vb, const bf16_t* __restrict__ Spt,
    const float* __restrict__ zp, bf16_t* __restrict__ ob) {
  const int bh = blockIdx.y, ch = blockIdx.x;
  const int b = bh >> 4, h = bh & 15;
  const bf16_t* Qc = qb + (((size_t)bh * SEQ) + (size_t)ch * CHUNK) * D_HEAD;
  const bf16_t* Kc = kb + (((size_t)bh * SEQ) + (size_t)ch * CHUNK) * D_HEAD;
  const bf16_t* Vc = vb + (((size_t)bh * SEQ) + (size_t)ch * CHUNK) * D_HEAD;
  const bf16_t* Sp = Spt + ((size_t)bh * NCHUNK + ch) * 4096;  // [e][d], ld 64
  const float* zpc = zp + ((size_t)bh * NCHUNK + ch) * 64;

  __shared__ __align__(16) bf16_t Qs[64 * 72];   // [t][d]
  __shared__ __align__(16) bf16_t Ks[64 * 72];   // [i][d]
  __shared__ __align__(16) bf16_t Scb[64 * 72];  // masked scores bf16 [t][i]
  __shared__ __align__(16) bf16_t Vt[64 * 72];   // [e][i]
  __shared__ float Sf[64 * 64];                  // masked scores fp32 (for den)
  __shared__ float den_s[64];

  const int tid = threadIdx.x, lane = tid & 31, wave = tid >> 5;
  {
    const int rr = tid >> 1, c0 = (tid & 1) * 32;
#pragma unroll 8
    for (int d = 0; d < 32; ++d) {
      Qs[rr * 72 + c0 + d] = Qc[rr * D_HEAD + c0 + d];
      Ks[rr * 72 + c0 + d] = Kc[rr * D_HEAD + c0 + d];
      Vt[(c0 + d) * 72 + rr] = Vc[rr * D_HEAD + c0 + d];
    }
  }
  __syncthreads();

  const int wt = wave >> 1, wi = wave & 1;  // 2x2 waves, each 32x32
  const int nL = lane & 15, mH = (lane >> 4) << 3;

  // scores = Q K^T (contraction over d; K rows are K-contiguous -> B operand direct)
  f32x8 sacc[2][2];
#pragma unroll
  for (int a = 0; a < 2; ++a)
#pragma unroll
    for (int c = 0; c < 2; ++c) sacc[a][c] = zero8();
  for (int d0 = 0; d0 < 64; d0 += 32) {
    bf16x16 af[2], bfr[2];
#pragma unroll
    for (int rt = 0; rt < 2; ++rt)
      af[rt] = frag_a(&Qs[(wt * 32 + rt * 16) * 72 + d0], 72, lane);
#pragma unroll
    for (int ct = 0; ct < 2; ++ct)
      bfr[ct] = frag_b(&Ks[(wi * 32 + ct * 16) * 72 + d0], 72, lane);
#pragma unroll
    for (int rt = 0; rt < 2; ++rt)
#pragma unroll
      for (int ct = 0; ct < 2; ++ct)
        sacc[rt][ct] = wmma_bf16(af[rt], bfr[ct], sacc[rt][ct]);
  }

  // causal mask (i <= t), stash fp32 + bf16 copies
#pragma unroll
  for (int rt = 0; rt < 2; ++rt)
#pragma unroll
    for (int ct = 0; ct < 2; ++ct)
#pragma unroll
      for (int r = 0; r < 8; ++r) {
        const int t = wt * 32 + rt * 16 + mH + r;
        const int i = wi * 32 + ct * 16 + nL;
        const float v = (i <= t) ? sacc[rt][ct][r] : 0.f;
        Scb[t * 72 + i] = (bf16_t)v;
        Sf[t * 64 + i] = v;
      }
  __syncthreads();

  // num = masked_scores @ V  +  Q @ S_prefix
  f32x8 oacc[2][2];
#pragma unroll
  for (int a = 0; a < 2; ++a)
#pragma unroll
    for (int c = 0; c < 2; ++c) oacc[a][c] = zero8();
  for (int i0 = 0; i0 < 64; i0 += 32) {
    bf16x16 af[2], bfr[2];
#pragma unroll
    for (int rt = 0; rt < 2; ++rt)
      af[rt] = frag_a(&Scb[(wt * 32 + rt * 16) * 72 + i0], 72, lane);
#pragma unroll
    for (int ct = 0; ct < 2; ++ct)
      bfr[ct] = frag_b(&Vt[(wi * 32 + ct * 16) * 72 + i0], 72, lane);
#pragma unroll
    for (int rt = 0; rt < 2; ++rt)
#pragma unroll
      for (int ct = 0; ct < 2; ++ct)
        oacc[rt][ct] = wmma_bf16(af[rt], bfr[ct], oacc[rt][ct]);
  }
  for (int d0 = 0; d0 < 64; d0 += 32) {
    bf16x16 af[2], bfr[2];
#pragma unroll
    for (int rt = 0; rt < 2; ++rt)
      af[rt] = frag_a(&Qs[(wt * 32 + rt * 16) * 72 + d0], 72, lane);
#pragma unroll
    for (int ct = 0; ct < 2; ++ct)
      bfr[ct] = frag_b(&Sp[(wi * 32 + ct * 16) * 64 + d0], 64, lane);
#pragma unroll
    for (int rt = 0; rt < 2; ++rt)
#pragma unroll
      for (int ct = 0; ct < 2; ++ct)
        oacc[rt][ct] = wmma_bf16(af[rt], bfr[ct], oacc[rt][ct]);
  }

  if (tid < 64) {
    const int t = tid;
    float den = 0.f;
    for (int i = 0; i <= t; ++i) den += Sf[t * 64 + i];
    for (int d = 0; d < 64; ++d) den += (float)Qs[t * 72 + d] * zpc[d];
    den_s[t] = fmaxf(den, 1e-6f);
  }
  __syncthreads();

#pragma unroll
  for (int rt = 0; rt < 2; ++rt)
#pragma unroll
    for (int ct = 0; ct < 2; ++ct)
#pragma unroll
      for (int r = 0; r < 8; ++r) {
        const int t = wt * 32 + rt * 16 + mH + r;
        const int e = wi * 32 + ct * 16 + nL;
        const float v = oacc[rt][ct][r] / den_s[t];
        const int l = ch * CHUNK + t;
        ob[((size_t)(b * SEQ + l)) * D_MODEL + h * D_HEAD + e] = (bf16_t)v;
      }
}

// ---------------------------------------------------------------------- host
extern "C" void kernel_launch(void* const* d_in, const int* in_sizes, int n_in,
                              void* d_out, int out_size, void* d_ws, size_t ws_size,
                              hipStream_t stream) {
  (void)in_sizes; (void)n_in; (void)out_size; (void)ws_size;
  const float* x    = (const float*)d_in[0];
  const float* wqkv = (const float*)d_in[1];
  const float* wout = (const float*)d_in[2];
  float* out = (float*)d_out;

  char* ws = (char*)d_ws;
  size_t off = 0;
  auto alloc = [&](size_t bytes) -> void* {
    void* p = (void*)(ws + off);
    off += (bytes + 255) & ~(size_t)255;
    return p;
  };

  const size_t ML = (size_t)BATCH * SEQ;  // 16384 rows
  bf16_t* xb    = (bf16_t*)alloc(ML * D_MODEL * 2);
  bf16_t* wqkvb = (bf16_t*)alloc((size_t)3 * D_MODEL * D_MODEL * 2);
  bf16_t* woutb = (bf16_t*)alloc((size_t)D_MODEL * D_MODEL * 2);
  bf16_t* qb    = (bf16_t*)alloc(ML * D_MODEL * 2);
  bf16_t* kb    = (bf16_t*)alloc(ML * D_MODEL * 2);
  bf16_t* vb    = (bf16_t*)alloc(ML * D_MODEL * 2);
  bf16_t* ob    = (bf16_t*)alloc(ML * D_MODEL * 2);
  float*  Sloc  = (float*)alloc((size_t)BH * NCHUNK * 64 * 64 * 4);
  float*  zloc  = (float*)alloc((size_t)BH * NCHUNK * 64 * 4);
  bf16_t* Spt   = (bf16_t*)alloc((size_t)BH * NCHUNK * 64 * 64 * 2);
  float*  zp    = (float*)alloc((size_t)BH * NCHUNK * 64 * 4);

  const int nx = (int)(ML * D_MODEL);
  const int nq = 3 * D_MODEL * D_MODEL;
  const int nw = D_MODEL * D_MODEL;
  cvt_bf16_k<<<dim3((nx / 4 + 255) / 256), 256, 0, stream>>>(x, xb, nx / 4);
  cvt_bf16_k<<<dim3((nq / 4 + 255) / 256), 256, 0, stream>>>(wqkv, wqkvb, nq / 4);
  cvt_bf16_k<<<dim3((nw / 4 + 255) / 256), 256, 0, stream>>>(wout, woutb, nw / 4);

  // QKV projection with fused phi + scatter epilogue
  gemm_bf16_k<1><<<dim3(3 * D_MODEL / 128, (int)(ML / 128)), 256, 0, stream>>>(
      xb, wqkvb, nullptr, qb, kb, vb, (int)ML, 3 * D_MODEL, D_MODEL);

  chunk_state_k<<<dim3(NCHUNK, BH), 128, 0, stream>>>(kb, vb, Sloc, zloc);
  prefix_k<<<dim3(BH), 256, 0, stream>>>(Sloc, zloc, Spt, zp);
  chunk_out_k<<<dim3(NCHUNK, BH), 128, 0, stream>>>(qb, kb, vb, Spt, zp, ob);

  // Output projection -> fp32 result
  gemm_bf16_k<0><<<dim3(D_MODEL / 128, (int)(ML / 128)), 256, 0, stream>>>(
      ob, woutb, out, nullptr, nullptr, nullptr, (int)ML, D_MODEL, D_MODEL);
}
